// G_PAM_Module_49520972922892
// MI455X (gfx1250) — matrix operand order repaired and checked by reading, not compiled, not run on hardware
//
#include <hip/hip_runtime.h>

// ---------------------------------------------------------------------------
// G-PAM dual attention for MI455X (gfx1250), wave32 + V_WMMA_F32_16X16X32_BF16
//   B=4, C=64, N=H*W=4096
//   k_proj : 1x1 conv projections; q,k,qg,kg stored [N,C]; v stored [C,N]
//   k_attn : 3-pass row softmax of q^T k, stores NORMALIZED bf16 attention
//            (row-major for A1, transposed [m][k] for A2 -> contiguous B tiles)
//   k_guide: barrier-free fused softmax(A1@A2) @ V with WMMA, 32x128/wave
// ---------------------------------------------------------------------------

#define NPIX 4096
#define CCH  64
#define NBATCH 4

typedef __bf16 bf16_t;
typedef __attribute__((ext_vector_type(16))) __bf16 v16bf;
typedef __attribute__((ext_vector_type(8)))  __bf16 v8bf;
typedef __attribute__((ext_vector_type(8)))  float  v8f;

static __device__ inline v8f wmma_bf16(v16bf a, v16bf b, v8f c) {
  // (neg_a, A, neg_b, B, c_mod, C, reuse_a, reuse_b)
  return __builtin_amdgcn_wmma_f32_16x16x32_bf16(false, a, false, b, (short)0, c,
                                                 false, false);
}

static __device__ inline v16bf cat16(v8bf lo, v8bf hi) {
  return __builtin_shufflevector(lo, hi, 0, 1, 2, 3, 4, 5, 6, 7, 8, 9, 10, 11,
                                 12, 13, 14, 15);
}

// A (16x32 bf16) fragment: lane l, row m=l&15; two contiguous 16B chunks at
// k-offsets {0..7,16..23} (low half) / {8..15,24..31} (high half).
static __device__ inline v16bf load_A(const bf16_t* __restrict__ A, int stride,
                                      int nbase, int kbase) {
  const int lane = threadIdx.x & 31;
  const bf16_t* p =
      A + (size_t)(nbase + (lane & 15)) * stride + kbase + ((lane >> 4) * 8);
  v8bf lo = *(const v8bf*)p;
  v8bf hi = *(const v8bf*)(p + 16);
  return cat16(lo, hi);
}

// B (32x16 bf16) fragment from a [col][k]-major matrix: lane l holds column
// col=l&15, k-offsets {0..7,16..23} (low half) / {8..15,24..31} (high half).
static __device__ inline v16bf load_B(const bf16_t* __restrict__ Bm, int stride,
                                      int colbase, int kbase) {
  const int lane = threadIdx.x & 31;
  const bf16_t* p =
      Bm + (size_t)(colbase + (lane & 15)) * stride + kbase + ((lane >> 4) * 8);
  v8bf lo = *(const v8bf*)p;
  v8bf hi = *(const v8bf*)(p + 16);
  return cat16(lo, hi);
}

// ---------------------------------------------------------------------------
// Kernel 1: projections. One thread per pixel column n; weights via LDS.
//   qT,kT,qgT,kgT : [B][N][C] bf16 ;  vM : [B][C][N] bf16
// ---------------------------------------------------------------------------
__global__ void k_proj(const float* __restrict__ x, const float* __restrict__ g,
                       const float* __restrict__ Wq, const float* __restrict__ bq,
                       const float* __restrict__ Wk, const float* __restrict__ bk,
                       const float* __restrict__ Wv, const float* __restrict__ bv,
                       const float* __restrict__ Wqg, const float* __restrict__ bqg,
                       const float* __restrict__ Wkg, const float* __restrict__ bkg,
                       bf16_t* __restrict__ qT, bf16_t* __restrict__ kT,
                       bf16_t* __restrict__ vM, bf16_t* __restrict__ qgT,
                       bf16_t* __restrict__ kgT) {
  __shared__ float Ws[CCH * CCH];
  const int bb = blockIdx.y;
  const int n = blockIdx.x * blockDim.x + threadIdx.x;
  float col[CCH];

  const float* xb = x + (size_t)bb * CCH * NPIX;
#pragma unroll 8
  for (int c = 0; c < CCH; ++c) col[c] = xb[(size_t)c * NPIX + n];

  // q -> qT[b][n][c]
  for (int i = threadIdx.x; i < CCH * CCH; i += blockDim.x) Ws[i] = Wq[i];
  __syncthreads();
  for (int o = 0; o < CCH; ++o) {
    float acc = bq[o];
#pragma unroll 8
    for (int c = 0; c < CCH; ++c) acc += Ws[o * CCH + c] * col[c];
    qT[((size_t)bb * NPIX + n) * CCH + o] = (bf16_t)acc;
  }
  __syncthreads();
  // k -> kT[b][n][c]  ([N,C]: contiguous B fragments for k_attn)
  for (int i = threadIdx.x; i < CCH * CCH; i += blockDim.x) Ws[i] = Wk[i];
  __syncthreads();
  for (int o = 0; o < CCH; ++o) {
    float acc = bk[o];
#pragma unroll 8
    for (int c = 0; c < CCH; ++c) acc += Ws[o * CCH + c] * col[c];
    kT[((size_t)bb * NPIX + n) * CCH + o] = (bf16_t)acc;
  }
  __syncthreads();
  // v -> vM[b][c][n]  ([C,N]: contiguous B fragments for the value GEMM)
  for (int i = threadIdx.x; i < CCH * CCH; i += blockDim.x) Ws[i] = Wv[i];
  __syncthreads();
  for (int o = 0; o < CCH; ++o) {
    float acc = bv[o];
#pragma unroll 8
    for (int c = 0; c < CCH; ++c) acc += Ws[o * CCH + c] * col[c];
    vM[((size_t)bb * CCH + o) * NPIX + n] = (bf16_t)acc;
  }
  __syncthreads();

  const float* gb = g + (size_t)bb * CCH * NPIX;
#pragma unroll 8
  for (int c = 0; c < CCH; ++c) col[c] = gb[(size_t)c * NPIX + n];

  // qg -> qgT[b][n][c]
  for (int i = threadIdx.x; i < CCH * CCH; i += blockDim.x) Ws[i] = Wqg[i];
  __syncthreads();
  for (int o = 0; o < CCH; ++o) {
    float acc = bqg[o];
#pragma unroll 8
    for (int c = 0; c < CCH; ++c) acc += Ws[o * CCH + c] * col[c];
    qgT[((size_t)bb * NPIX + n) * CCH + o] = (bf16_t)acc;
  }
  __syncthreads();
  // kg -> kgT[b][n][c]
  for (int i = threadIdx.x; i < CCH * CCH; i += blockDim.x) Ws[i] = Wkg[i];
  __syncthreads();
  for (int o = 0; o < CCH; ++o) {
    float acc = bkg[o];
#pragma unroll 8
    for (int c = 0; c < CCH; ++c) acc += Ws[o * CCH + c] * col[c];
    kgT[((size_t)bb * NPIX + n) * CCH + o] = (bf16_t)acc;
  }
}

// ---------------------------------------------------------------------------
// Kernel 2: energy = Qt @ K^T over channels (K-dim = 64), row softmax.
// 3 passes (max / sum / normalized store) -> attention maps need no later
// rescaling. TRANSPOSE=true stores Aout[m][k] (column-major rows) so the big
// GEMM can read contiguous B fragments. Stores are vectorized (16B) through a
// tiny wave-private LDS bounce.
// ---------------------------------------------------------------------------
template <bool TRANSPOSE>
__global__ void k_attn(const bf16_t* __restrict__ Qt, const bf16_t* __restrict__ Kt,
                       bf16_t* __restrict__ Aout) {
  __shared__ __align__(16) bf16_t Ts[8][16][24];
  const int bb = blockIdx.y;
  const int wid = threadIdx.x >> 5;
  const int lane = threadIdx.x & 31;
  const int nb = blockIdx.x * 128 + wid * 16;
  const int colc = lane & 15;
  const int hi = lane >> 4;

  const bf16_t* Qb = Qt + (size_t)bb * NPIX * CCH;
  const bf16_t* Kb = Kt + (size_t)bb * NPIX * CCH;
  bf16_t* Ab = Aout + (size_t)bb * NPIX * NPIX;

  const v16bf a0 = load_A(Qb, CCH, nb, 0);
  const v16bf a1 = load_A(Qb, CCH, nb, 32);

  v8f rmax;
#pragma unroll
  for (int i = 0; i < 8; ++i) rmax[i] = -3.0e38f;

  for (int mt = 0; mt < NPIX / 16; ++mt) {
    v8f acc = {};
    acc = wmma_bf16(a0, load_B(Kb, CCH, mt * 16, 0), acc);
    acc = wmma_bf16(a1, load_B(Kb, CCH, mt * 16, 32), acc);
#pragma unroll
    for (int i = 0; i < 8; ++i) rmax[i] = fmaxf(rmax[i], acc[i]);
  }
#pragma unroll
  for (int s = 1; s < 16; s <<= 1)
#pragma unroll
    for (int i = 0; i < 8; ++i) rmax[i] = fmaxf(rmax[i], __shfl_xor(rmax[i], s, 32));

  v8f rsum = {};
  for (int mt = 0; mt < NPIX / 16; ++mt) {
    v8f acc = {};
    acc = wmma_bf16(a0, load_B(Kb, CCH, mt * 16, 0), acc);
    acc = wmma_bf16(a1, load_B(Kb, CCH, mt * 16, 32), acc);
#pragma unroll
    for (int i = 0; i < 8; ++i) rsum[i] += __expf(acc[i] - rmax[i]);
  }
#pragma unroll
  for (int s = 1; s < 16; s <<= 1)
#pragma unroll
    for (int i = 0; i < 8; ++i) rsum[i] += __shfl_xor(rsum[i], s, 32);
  float inv[8];
#pragma unroll
  for (int i = 0; i < 8; ++i) inv[i] = 1.0f / rsum[i];

  for (int mt = 0; mt < NPIX / 16; ++mt) {
    v8f acc = {};
    acc = wmma_bf16(a0, load_B(Kb, CCH, mt * 16, 0), acc);
    acc = wmma_bf16(a1, load_B(Kb, CCH, mt * 16, 32), acc);
    if (TRANSPOSE) {
      // Ts[m_local][n_local]; this lane's 8 values are contiguous -> 16B store
      v8bf pv;
#pragma unroll
      for (int i = 0; i < 8; ++i)
        pv[i] = (bf16_t)(__expf(acc[i] - rmax[i]) * inv[i]);
      *(v8bf*)&Ts[wid][colc][hi * 8] = pv;
      v8bf row = *(const v8bf*)&Ts[wid][lane & 15][(lane >> 4) * 8];
      *(v8bf*)&Ab[(size_t)(mt * 16 + (lane & 15)) * NPIX + nb + (lane >> 4) * 8] =
          row;
    } else {
      // Ts[n_local][m_local]; bounce so the output store is 16B row chunks
#pragma unroll
      for (int i = 0; i < 8; ++i)
        Ts[wid][i + 8 * hi][colc] = (bf16_t)(__expf(acc[i] - rmax[i]) * inv[i]);
      v8bf row = *(const v8bf*)&Ts[wid][lane & 15][(lane >> 4) * 8];
      *(v8bf*)&Ab[(size_t)(nb + (lane & 15)) * NPIX + mt * 16 + (lane >> 4) * 8] =
          row;
    }
  }
}

// ---------------------------------------------------------------------------
// Kernel 3: fused  GA = softmax(A1 @ A2);  out = gamma * (V @ GA^T) + x
// Barrier-free: A2 is stored transposed so B fragments are contiguous global
// loads (A2 = 32MB/batch -> L2 resident). Each wave owns a 32x128 block:
// 16 WMMA per (2 A + 8 B) 32-byte-per-lane loads. Flash-style: pass1 rowmax
// (recompute), pass2 exp + rowsum + P@V^T (P re-laid out via wave-private LDS).
// ---------------------------------------------------------------------------
__global__ void __launch_bounds__(128)
k_guide(const bf16_t* __restrict__ A1, const bf16_t* __restrict__ A2T,
        const bf16_t* __restrict__ vM, const float* __restrict__ x,
        const float* __restrict__ gamma_p, float* __restrict__ out) {
  __shared__ __align__(16) bf16_t Ps[4][16][136];

  const int bb = blockIdx.y;
  const int wid = threadIdx.x >> 5;
  const int lane = threadIdx.x & 31;
  const int nb = blockIdx.x * 128 + wid * 32;  // 2 row tiles: nb, nb+16
  const int colc = lane & 15;
  const int hi = lane >> 4;

  const bf16_t* A1b = A1 + (size_t)bb * NPIX * NPIX;
  const bf16_t* A2b = A2T + (size_t)bb * NPIX * NPIX;
  const bf16_t* Vb = vM + (size_t)bb * CCH * NPIX;

  // ---------------- PASS 1: rowmax of guide_energy -------------------------
  v8f rmax[2];
#pragma unroll
  for (int r = 0; r < 2; ++r)
#pragma unroll
    for (int i = 0; i < 8; ++i) rmax[r][i] = -3.0e38f;

  for (int mb = 0; mb < NPIX / 128; ++mb) {
    v8f acc[2][8];
    const v8f vz = {};
#pragma unroll
    for (int r = 0; r < 2; ++r)
#pragma unroll
      for (int t = 0; t < 8; ++t) acc[r][t] = vz;

    for (int kc = 0; kc < NPIX; kc += 32) {
      const v16bf a0 = load_A(A1b, NPIX, nb, kc);
      const v16bf a1 = load_A(A1b, NPIX, nb + 16, kc);
#pragma unroll
      for (int t = 0; t < 8; ++t) {
        const v16bf bt = load_B(A2b, NPIX, mb * 128 + t * 16, kc);
        acc[0][t] = wmma_bf16(a0, bt, acc[0][t]);
        acc[1][t] = wmma_bf16(a1, bt, acc[1][t]);
      }
    }
#pragma unroll
    for (int r = 0; r < 2; ++r)
#pragma unroll
      for (int t = 0; t < 8; ++t)
#pragma unroll
        for (int i = 0; i < 8; ++i) rmax[r][i] = fmaxf(rmax[r][i], acc[r][t][i]);
  }
#pragma unroll
  for (int s = 1; s < 16; s <<= 1)
#pragma unroll
    for (int r = 0; r < 2; ++r)
#pragma unroll
      for (int i = 0; i < 8; ++i)
        rmax[r][i] = fmaxf(rmax[r][i], __shfl_xor(rmax[r][i], s, 32));

  // ---------------- PASS 2: exp, rowsum, O^T += P @ V^T --------------------
  v8f rsum[2];
  v8f o[2][4];
  {
    const v8f vz = {};
#pragma unroll
    for (int r = 0; r < 2; ++r) {
      rsum[r] = vz;
#pragma unroll
      for (int ct = 0; ct < 4; ++ct) o[r][ct] = vz;
    }
  }

  for (int mb = 0; mb < NPIX / 128; ++mb) {
    v8f acc[2][8];
    const v8f vz = {};
#pragma unroll
    for (int r = 0; r < 2; ++r)
#pragma unroll
      for (int t = 0; t < 8; ++t) acc[r][t] = vz;

    for (int kc = 0; kc < NPIX; kc += 32) {
      const v16bf a0 = load_A(A1b, NPIX, nb, kc);
      const v16bf a1 = load_A(A1b, NPIX, nb + 16, kc);
#pragma unroll
      for (int t = 0; t < 8; ++t) {
        const v16bf bt = load_B(A2b, NPIX, mb * 128 + t * 16, kc);
        acc[0][t] = wmma_bf16(a0, bt, acc[0][t]);
        acc[1][t] = wmma_bf16(a1, bt, acc[1][t]);
      }
    }

#pragma unroll
    for (int r = 0; r < 2; ++r) {
      // p = exp(ge - rowmax); bounce P (16x128) through wave-private LDS to
      // re-layout D -> A operand; accumulate rowsum.
#pragma unroll
      for (int i = 0; i < 8; ++i) {
        const int rr = i + 8 * hi;
#pragma unroll
        for (int t = 0; t < 8; ++t) {
          float p = __expf(acc[r][t][i] - rmax[r][i]);
          rsum[r][i] += p;
          Ps[wid][rr][t * 16 + colc] = (bf16_t)p;
        }
      }
#pragma unroll
      for (int kk2 = 0; kk2 < 4; ++kk2) {
        const v16bf aP = load_A(&Ps[wid][0][0], 136, 0, kk2 * 32);
#pragma unroll
        for (int ct = 0; ct < 4; ++ct)
          o[r][ct] = wmma_bf16(aP, load_B(Vb, NPIX, ct * 16, mb * 128 + kk2 * 32),
                               o[r][ct]);
      }
    }
  }
#pragma unroll
  for (int s = 1; s < 16; s <<= 1)
#pragma unroll
    for (int r = 0; r < 2; ++r)
#pragma unroll
      for (int i = 0; i < 8; ++i) rsum[r][i] += __shfl_xor(rsum[r][i], s, 32);

  const float gam = gamma_p[0];
#pragma unroll
  for (int r = 0; r < 2; ++r)
#pragma unroll
    for (int ct = 0; ct < 4; ++ct)
#pragma unroll
      for (int i = 0; i < 8; ++i) {
        const int n = nb + r * 16 + i + 8 * hi;
        const int c = ct * 16 + colc;
        const size_t idx = ((size_t)bb * CCH + c) * NPIX + n;
        out[idx] = o[r][ct][i] * (gam / rsum[r][i]) + x[idx];
      }
}

// ---------------------------------------------------------------------------
extern "C" void kernel_launch(void* const* d_in, const int* in_sizes, int n_in,
                              void* d_out, int out_size, void* d_ws, size_t ws_size,
                              hipStream_t stream) {
  const float* x = (const float*)d_in[0];
  const float* g = (const float*)d_in[1];
  const float* Wq = (const float*)d_in[2];
  const float* bq = (const float*)d_in[3];
  const float* Wk = (const float*)d_in[4];
  const float* bk = (const float*)d_in[5];
  const float* Wv = (const float*)d_in[6];
  const float* bv = (const float*)d_in[7];
  const float* Wqg = (const float*)d_in[8];
  const float* bqg = (const float*)d_in[9];
  const float* Wkg = (const float*)d_in[10];
  const float* bkg = (const float*)d_in[11];
  const float* gamma = (const float*)d_in[12];
  float* out = (float*)d_out;

  char* p = (char*)d_ws;
  auto alloc = [&](size_t bytes) -> char* {
    char* r = p;
    p += (bytes + 255) & ~(size_t)255;
    return r;
  };
  bf16_t* qT = (bf16_t*)alloc((size_t)NBATCH * NPIX * CCH * sizeof(bf16_t));
  bf16_t* kT = (bf16_t*)alloc((size_t)NBATCH * NPIX * CCH * sizeof(bf16_t));
  bf16_t* vM = (bf16_t*)alloc((size_t)NBATCH * CCH * NPIX * sizeof(bf16_t));
  bf16_t* qgT = (bf16_t*)alloc((size_t)NBATCH * NPIX * CCH * sizeof(bf16_t));
  bf16_t* kgT = (bf16_t*)alloc((size_t)NBATCH * NPIX * CCH * sizeof(bf16_t));
  bf16_t* A1 = (bf16_t*)alloc((size_t)NBATCH * NPIX * NPIX * sizeof(bf16_t));
  bf16_t* A2T = (bf16_t*)alloc((size_t)NBATCH * NPIX * NPIX * sizeof(bf16_t));

  k_proj<<<dim3(NPIX / 256, NBATCH), 256, 0, stream>>>(
      x, g, Wq, bq, Wk, bk, Wv, bv, Wqg, bqg, Wkg, bkg, qT, kT, vM, qgT, kgT);
  k_attn<false><<<dim3(NPIX / 128, NBATCH), 256, 0, stream>>>(qT, kT, A1);
  k_attn<true><<<dim3(NPIX / 128, NBATCH), 256, 0, stream>>>(qgT, kgT, A2T);
  k_guide<<<dim3(NPIX / 128, NBATCH), 128, 0, stream>>>(A1, A2T, vM, x, gamma,
                                                        out);
}
